// HardThresholdingMultiheadAttention_55456617726458
// MI455X (gfx1250) — compile-verified, matrix operand order checked by reading, not run on hardware
//
#include <hip/hip_runtime.h>
#include <hip/hip_bf16.h>
#include <cstdint>
#include <cstddef>

typedef __attribute__((ext_vector_type(16))) _Float16 v16h;
typedef __attribute__((ext_vector_type(8)))  float    v8f;

#define B_   2
#define S_   4096
#define HID_ 512
#define NH_  8
#define HD_  64

constexpr int   SC_STRIDE = 4100;     // padded LDS row stride (floats), avoids bank conflicts
constexpr float SM_SCALE  = 0.125f;   // 1/sqrt(64)

// ---------------------------------------------------------------------------
// WMMA fragment helpers (wave32, v_wmma_f32_16x16x32_f16)
// A-frag (16x32, 16-bit): lane L -> row = L&15, half = L>>4;
//   elements e<8  : K = half*8 + e
//   elements e>=8 : K = 16 + half*8 + (e-8)
// B-frag assumed mirrored with lane -> column N = L&15, same K pattern.
// C/D (16x16 f32): element i -> row = i + 8*(L>>4), col = L&15.
// ---------------------------------------------------------------------------

__device__ __forceinline__ v8f wmma_f16(v16h a, v16h b, v8f c) {
  return __builtin_amdgcn_wmma_f32_16x16x32_f16(
      /*neg_a=*/false, a, /*neg_b=*/false, b,
      /*c_mod=*/(short)0, c, /*reuse_a=*/false, /*reuse_b=*/false);
}

// Row-major fp32 source: element (r, k) at base[r*ld + k], r = lane&15.
__device__ __forceinline__ v16h frag_rowmajor_f32(const float* __restrict__ base, int ld) {
  const int lane = threadIdx.x & 31;
  const int r    = lane & 15;
  const int half = lane >> 4;
  const float* p = base + (size_t)r * ld + half * 8;
  v16h f;
#pragma unroll
  for (int e = 0; e < 8; ++e) f[e] = (_Float16)p[e];
#pragma unroll
  for (int e = 0; e < 8; ++e) f[8 + e] = (_Float16)p[16 + e];
  return f;
}

// Row-major f16 source.
__device__ __forceinline__ v16h frag_rowmajor_f16(const _Float16* __restrict__ base, int ld) {
  const int lane = threadIdx.x & 31;
  const int r    = lane & 15;
  const int half = lane >> 4;
  const _Float16* p = base + (size_t)r * ld + half * 8;
  v16h f;
#pragma unroll
  for (int e = 0; e < 8; ++e) f[e] = p[e];
#pragma unroll
  for (int e = 0; e < 8; ++e) f[8 + e] = p[16 + e];
  return f;
}

// Column-fixed f16 source: B[k][n] at base[k*ld + n], n = lane&15, k = pattern.
__device__ __forceinline__ v16h frag_colmajor_f16(const _Float16* __restrict__ base, int ld) {
  const int lane = threadIdx.x & 31;
  const int c    = lane & 15;
  const int half = lane >> 4;
  const _Float16* p = base + c + (size_t)(half * 8) * ld;
  v16h f;
#pragma unroll
  for (int e = 0; e < 8; ++e) f[e] = p[(size_t)e * ld];
#pragma unroll
  for (int e = 0; e < 8; ++e) f[8 + e] = p[(size_t)(16 + e) * ld];
  return f;
}

// ---------------------------------------------------------------------------
// Kernel 1: projection  Y = X @ W^T + b, fp32 in, f16 out in [b,h,s,d] layout.
// X: (B*S, HID) fp32, W: (HID, HID) fp32, Y: f16 [B][NH][S][HD].
// One 16x16 output tile per wave; 8 waves per block; 16384 tiles total.
// ---------------------------------------------------------------------------
__global__ __launch_bounds__(256) void mha_proj_kernel(
    const float* __restrict__ X, const float* __restrict__ W,
    const float* __restrict__ bias, _Float16* __restrict__ Y) {
  const int wave = threadIdx.x >> 5;
  const int lane = threadIdx.x & 31;
  const int t    = blockIdx.x * 8 + wave;    // tile id, [0, 16384)
  const int mt   = t >> 5;                   // 512 row tiles
  const int nt   = t & 31;                   // 32 col tiles
  const int m0   = mt * 16;
  const int n0   = nt * 16;

  v8f acc = {};
#pragma unroll 4
  for (int k0 = 0; k0 < HID_; k0 += 32) {
    v16h a = frag_rowmajor_f32(X + (size_t)m0 * HID_ + k0, HID_);
    v16h b = frag_rowmajor_f32(W + (size_t)n0 * HID_ + k0, HID_);   // B[k][n] = W[n][k]
    acc = wmma_f16(a, b, acc);
  }

  const int half = lane >> 4;
  const int col  = lane & 15;
  const int n    = n0 + col;
  const float bn = bias[n];
  const int h = n >> 6;         // n / HD
  const int d = n & 63;         // n % HD
#pragma unroll
  for (int i = 0; i < 8; ++i) {
    const int m  = m0 + i + half * 8;
    const int bb = m >> 12;     // m / S
    const int s  = m & 4095;    // m % S
    Y[(((size_t)(bb * NH_ + h) * S_) + s) * HD_ + d] = (_Float16)(acc[i] + bn);
  }
}

// ---------------------------------------------------------------------------
// Kernel 2: attention for one (b, h, 16-query-row block).
// LDS: scores 16 x SC_STRIDE fp32 (~256 KB, fits CDNA5 320 KB/WGP) + 4 KB partials.
// ---------------------------------------------------------------------------
__global__ __launch_bounds__(256) void mha_attn_kernel(
    const _Float16* __restrict__ Qh, const _Float16* __restrict__ Kh,
    const _Float16* __restrict__ Vh, const unsigned char* __restrict__ kpm,
    float* __restrict__ Out, float* __restrict__ Wt) {
  extern __shared__ float smem[];
  float* sc  = smem;                         // [16][SC_STRIDE]
  float* red = smem + 16 * SC_STRIDE;        // [4][256] partial C tiles

  const int wave = threadIdx.x >> 5;
  const int lane = threadIdx.x & 31;
  const int half = lane >> 4;
  const int col  = lane & 15;

  const int nqt = S_ / 16;                   // 256
  const int qt  = blockIdx.x % nqt;
  const int h   = (blockIdx.x / nqt) % NH_;
  const int b   = blockIdx.x / (nqt * NH_);
  const int q0  = qt * 16;

  const size_t headOff = (size_t)(b * NH_ + h) * S_ * HD_;
  const _Float16* Q = Qh + headOff;
  const _Float16* K = Kh + headOff;
  const _Float16* V = Vh + headOff;

  // ---- Phase 1: scores = Q K^T * scale, masked, into LDS -------------------
  const v16h aq0 = frag_rowmajor_f16(Q + (size_t)q0 * HD_ + 0, HD_);
  const v16h aq1 = frag_rowmajor_f16(Q + (size_t)q0 * HD_ + 32, HD_);
  for (int ct = wave; ct < nqt; ct += 8) {
    const int n0 = ct * 16;
    v16h bk0 = frag_rowmajor_f16(K + (size_t)n0 * HD_ + 0, HD_);    // B[d][n]=K[n][d]
    v16h bk1 = frag_rowmajor_f16(K + (size_t)n0 * HD_ + 32, HD_);
    v8f acc = {};
    acc = wmma_f16(aq0, bk0, acc);
    acc = wmma_f16(aq1, bk1, acc);
    const int skey  = n0 + col;
    const bool msk  = kpm[b * S_ + skey] != 0;
#pragma unroll
    for (int i = 0; i < 8; ++i) {
      const int r = i + half * 8;
      sc[r * SC_STRIDE + skey] = msk ? -__builtin_inff() : acc[i] * SM_SCALE;
    }
  }
  __syncthreads();

  // ---- Phase 2: per-row softmax; write normalized weights to global --------
#pragma unroll
  for (int rr = 0; rr < 2; ++rr) {
    const int r = wave * 2 + rr;
    float m = -__builtin_inff();
    for (int j = lane; j < S_; j += 32) m = fmaxf(m, sc[r * SC_STRIDE + j]);
#pragma unroll
    for (int o = 16; o > 0; o >>= 1) m = fmaxf(m, __shfl_xor(m, o, 32));
    float sum = 0.f;
    for (int j = lane; j < S_; j += 32) {
      const float e = __expf(sc[r * SC_STRIDE + j] - m);
      sc[r * SC_STRIDE + j] = e;
      sum += e;
    }
#pragma unroll
    for (int o = 16; o > 0; o >>= 1) sum += __shfl_xor(sum, o, 32);
    const float inv = 1.f / sum;
    float* wrow = Wt + ((size_t)(b * NH_ + h) * S_ + (q0 + r)) * (size_t)S_;
    for (int j = lane; j < S_; j += 32) {
      const float w = sc[r * SC_STRIDE + j] * inv;
      sc[r * SC_STRIDE + j] = w;
      wrow[j] = w;                            // coalesced 1.07 GB stream: the roofline term
    }
  }
  __syncthreads();

  // ---- Phase 3: out = W @ V (K split over wave pairs) ----------------------
  const int ntile = wave & 3;                 // which 16-wide slice of HD
  const int khalf = wave >> 2;                // K range half
  v8f acc = {};
  const int kbeg = khalf * (S_ / 2);
  for (int k0 = kbeg; k0 < kbeg + S_ / 2; k0 += 32) {
    v16h a  = frag_rowmajor_f32(sc + k0, SC_STRIDE);                  // weights fp32->f16
    v16h bv = frag_colmajor_f16(V + (size_t)k0 * HD_ + ntile * 16, HD_);
    acc = wmma_f16(a, bv, acc);
  }
  if (khalf == 1) {
#pragma unroll
    for (int i = 0; i < 8; ++i) red[ntile * 256 + lane * 8 + i] = acc[i];
  }
  __syncthreads();
  if (khalf == 0) {
#pragma unroll
    for (int i = 0; i < 8; ++i) acc[i] += red[ntile * 256 + lane * 8 + i];
#pragma unroll
    for (int i = 0; i < 8; ++i) {
      const int r = i + half * 8;
      Out[((size_t)b * S_ + (q0 + r)) * HID_ + h * HD_ + ntile * 16 + col] = acc[i];
    }
  }
}

// ---------------------------------------------------------------------------
extern "C" void kernel_launch(void* const* d_in, const int* in_sizes, int n_in,
                              void* d_out, int out_size, void* d_ws, size_t ws_size,
                              hipStream_t stream) {
  (void)in_sizes; (void)n_in; (void)out_size; (void)ws_size;
  const float* query = (const float*)d_in[0];
  const float* key   = (const float*)d_in[1];
  const float* value = (const float*)d_in[2];
  const unsigned char* kpm = (const unsigned char*)d_in[3];   // jnp bool -> 1 byte
  const float* Wq = (const float*)d_in[4];
  const float* bq = (const float*)d_in[5];
  const float* Wk = (const float*)d_in[6];
  const float* bk = (const float*)d_in[7];
  const float* Wv = (const float*)d_in[8];
  const float* bv = (const float*)d_in[9];

  const size_t per = (size_t)B_ * NH_ * S_ * HD_;   // 4.19M f16 each
  _Float16* qh = (_Float16*)d_ws;
  _Float16* kh = qh + per;
  _Float16* vh = kh + per;

  float* out = (float*)d_out;                        // (B, S, HID)
  float* wts = out + (size_t)B_ * S_ * HID_;         // (B, NH, S, S)

  // Projections: 16384 tiles / 8 waves per block = 2048 blocks.
  mha_proj_kernel<<<2048, 256, 0, stream>>>(query, Wq, bq, qh);
  mha_proj_kernel<<<2048, 256, 0, stream>>>(key,   Wk, bk, kh);
  mha_proj_kernel<<<2048, 256, 0, stream>>>(value, Wv, bv, vh);

  // Attention: one block per (b, h, 16 q-rows) = 2*8*256 = 4096 blocks.
  const size_t smem_bytes = (size_t)(16 * SC_STRIDE + 4 * 256) * sizeof(float);
  mha_attn_kernel<<<B_ * NH_ * (S_ / 16), 256, smem_bytes, stream>>>(
      qh, kh, vh, kpm, out, wts);
}
